// iot_37452114821339
// MI455X (gfx1250) — matrix operand order repaired and checked by reading, not compiled, not run on hardware
//
#include <hip/hip_runtime.h>
#include <stdint.h>

typedef __attribute__((ext_vector_type(16))) _Float16 v16h;
typedef __attribute__((ext_vector_type(8)))  float    v8f;

#define S     8192
#define D     128
#define NH    4096        // S/2
#define EPS   0.1f
#define TILES 512         // S/16

// ---- bf16 helpers (conversion is just a shift; why bf16 is the right K format) ----
__device__ __forceinline__ float bf_lo(unsigned u) { return __uint_as_float(u << 16); }
__device__ __forceinline__ float bf_hi(unsigned u) { return __uint_as_float(u & 0xffff0000u); }
__device__ __forceinline__ unsigned short f2bf(float f) {
    unsigned u = __float_as_uint(f);
    return (unsigned short)((u + 0x7fffu + ((u >> 16) & 1u)) >> 16); // RNE
}

// ---- phase 0: b = ones, maxcost = 0 ----
__global__ void iot_init_kernel(float* __restrict__ b, unsigned* __restrict__ maxcost) {
    int i = blockIdx.x * blockDim.x + threadIdx.x;
    if (i < S) b[i] = 1.0f;
    if (i == 0) *maxcost = 0u;
}

// ---- phase 1: L2-normalize rows of z, emit f16 for the WMMA GEMM ----
__global__ __launch_bounds__(128) void iot_normalize_kernel(const float* __restrict__ z,
                                                            _Float16* __restrict__ znh) {
    const int row = blockIdx.x;
    const int t   = threadIdx.x;          // 0..127
    float v  = z[(size_t)row * D + t];
    float ss = v * v;
    #pragma unroll
    for (int o = 16; o; o >>= 1) ss += __shfl_xor(ss, o, 32);
    __shared__ float w4[4];
    if ((t & 31) == 0) w4[t >> 5] = ss;
    __syncthreads();
    float tot = w4[0] + w4[1] + w4[2] + w4[3];
    znh[(size_t)row * D + t] = (_Float16)(v * (1.0f / sqrtf(tot)));
}

// ---- phase 2: K = exp((cos-1)/eps) in bf16 via v_wmma_f32_16x16x32_f16; track max cost ----
// One wave computes one 16x16 tile of Z * Z^T (K-loop of 4 WMMAs over D=128).
__global__ __launch_bounds__(256) void iot_gemm_exp_kernel(const _Float16* __restrict__ znh,
                                                           unsigned short* __restrict__ K,
                                                           unsigned* __restrict__ maxcost) {
    const int lane = threadIdx.x;              // 0..31
    const int wave = threadIdx.y;              // 0..7
    const int tile = blockIdx.x * 8 + wave;    // 0..262143
    const int tr = tile / TILES, tc = tile % TILES;
    const int m  = lane & 15;
    const int kh = lane >> 4;

    const _Float16* rowA = znh + (size_t)(tr * 16 + m) * D;
    const _Float16* rowB = znh + (size_t)(tc * 16 + m) * D;

    v8f acc = {};
    #pragma unroll
    for (int k0 = 0; k0 < D; k0 += 32) {
        union { v16h v; uint4 q[2]; } A, B;
        // A 16x32 f16 layout: lane holds row m; halves 0..7 -> K = k0+8*kh..+7,
        //                     halves 8..15 -> K = k0+16+8*kh..+7
        A.q[0] = *(const uint4*)(rowA + k0 + kh * 8);
        A.q[1] = *(const uint4*)(rowA + k0 + 16 + kh * 8);
        // B 32x16 f16 layout: lane holds col m; halves 0..7 -> K = k0+8*kh..+7, halves 8..15 -> K = k0+16+8*kh..+7
        B.q[0] = *(const uint4*)(rowB + k0 + kh * 8);
        B.q[1] = *(const uint4*)(rowB + k0 + 16 + kh * 8);
        acc = __builtin_amdgcn_wmma_f32_16x16x32_f16(false, A.v, false, B.v,
                                                     (short)0, acc, false, false);
    }

    // C layout: VGPR r -> row (tr*16 + r + 8*kh), col (tc*16 + m)
    float mc = 0.f;
    const int col = tc * 16 + m;
    unsigned short* outp = K + (size_t)(tr * 16 + kh * 8) * S + col;
    #pragma unroll
    for (int r = 0; r < 8; ++r) {
        float cosv = acc[r];
        float cost = fmaxf(1.0f - cosv, 0.0f);   // clamp keeps uint-bit atomicMax valid
        mc = fmaxf(mc, cost);
        outp[(size_t)r * S] = f2bf(__expf(-cost * (1.0f / EPS)));
    }

    __shared__ float red[256];
    const int tid = wave * 32 + lane;
    red[tid] = mc;
    __syncthreads();
    for (int off = 128; off; off >>= 1) {
        if (tid < off) red[tid] = fmaxf(red[tid], red[tid + off]);
        __syncthreads();
    }
    if (tid == 0) atomicMax(maxcost, __float_as_uint(red[0]));  // non-negative floats: bit order == value order
}

// ---- phase 3: diagonal gets the max cost ----
__global__ void iot_diag_kernel(unsigned short* __restrict__ K, const unsigned* __restrict__ maxcost) {
    int i = blockIdx.x * blockDim.x + threadIdx.x;
    if (i < S) {
        float C = __uint_as_float(*maxcost);
        K[(size_t)i * S + i] = f2bf(__expf(-C * (1.0f / EPS)));
    }
}

// ---- phase 4: Sinkhorn step: out = 1 / (K x). One 256-thread block per row; streams 16KB/row ----
__global__ __launch_bounds__(256) void iot_matvec_recip_kernel(const unsigned short* __restrict__ K,
                                                               const float* __restrict__ x,
                                                               float* __restrict__ out) {
    const size_t row = blockIdx.x;
    const uint4*  rowp = (const uint4*)(K + row * S);   // 8 bf16 per uint4; 1024 per row
    const float4* x4   = (const float4*)x;
    const int tid = threadIdx.x;
    float acc = 0.f;
    #pragma unroll
    for (int it = 0; it < 4; ++it) {
        int q = it * 256 + tid;
        uint4  u  = rowp[q];
        float4 xa = x4[2 * q], xb = x4[2 * q + 1];
        acc = fmaf(bf_lo(u.x), xa.x, acc);
        acc = fmaf(bf_hi(u.x), xa.y, acc);
        acc = fmaf(bf_lo(u.y), xa.z, acc);
        acc = fmaf(bf_hi(u.y), xa.w, acc);
        acc = fmaf(bf_lo(u.z), xb.x, acc);
        acc = fmaf(bf_hi(u.z), xb.y, acc);
        acc = fmaf(bf_lo(u.w), xb.z, acc);
        acc = fmaf(bf_hi(u.w), xb.w, acc);
    }
    #pragma unroll
    for (int o = 16; o; o >>= 1) acc += __shfl_xor(acc, o, 32);
    __shared__ float wsum[8];
    if ((tid & 31) == 0) wsum[tid >> 5] = acc;
    __syncthreads();
    if (tid == 0) {
        float s = 0.f;
        #pragma unroll
        for (int w = 0; w < 8; ++w) s += wsum[w];
        out[row] = 1.0f / s;
    }
}

// ---- phase 5: L = -(1/S) * sum_t log( a[t] * K[t, (t+NH)%S] * b[(t+NH)%S] ) ----
__global__ __launch_bounds__(1024) void iot_loss_kernel(const unsigned short* __restrict__ K,
                                                        const float* __restrict__ a,
                                                        const float* __restrict__ b,
                                                        float* __restrict__ out) {
    const int tid = threadIdx.x;
    float s = 0.f;
    for (int t = tid; t < S; t += 1024) {
        int c = (t + NH) & (S - 1);
        float kv = __uint_as_float(((unsigned)K[(size_t)t * S + c]) << 16);
        s += logf(a[t] * kv * b[c]);
    }
    #pragma unroll
    for (int o = 16; o; o >>= 1) s += __shfl_xor(s, o, 32);
    __shared__ float w32[32];
    if ((tid & 31) == 0) w32[tid >> 5] = s;
    __syncthreads();
    if (tid == 0) {
        float tot = 0.f;
        #pragma unroll
        for (int w = 0; w < 32; ++w) tot += w32[w];
        out[0] = -tot / (float)S;
    }
}

extern "C" void kernel_launch(void* const* d_in, const int* in_sizes, int n_in,
                              void* d_out, int out_size, void* d_ws, size_t ws_size,
                              hipStream_t stream) {
    (void)in_sizes; (void)n_in; (void)out_size; (void)ws_size;
    const float* z = (const float*)d_in[0];

    // workspace layout: K bf16 (128 MiB) | zn f16 (2 MiB) | 3 vectors | maxcost scalar
    char* ws = (char*)d_ws;
    unsigned short* Km = (unsigned short*)ws;
    size_t offK = (size_t)S * S * sizeof(unsigned short);
    _Float16* znh = (_Float16*)(ws + offK);
    size_t offZ = offK + (size_t)S * D * sizeof(_Float16);
    float* V0 = (float*)(ws + offZ);
    float* V1 = V0 + S;
    float* V2 = V1 + S;
    unsigned* maxcost = (unsigned*)(V2 + S);

    iot_init_kernel<<<S / 256, 256, 0, stream>>>(V0, maxcost);
    iot_normalize_kernel<<<S, 128, 0, stream>>>(z, znh);
    iot_gemm_exp_kernel<<<(TILES * TILES) / 8, dim3(32, 8), 0, stream>>>(znh, Km, maxcost);
    iot_diag_kernel<<<S / 256, 256, 0, stream>>>(Km, maxcost);

    // P = diag(a) K diag(b); per reference iteration: a = 1/(K b); b = 1/(K a); (a,b) <- (b,a)
    float *pb = V0, *pt = V1, *pa = V2;
    for (int it = 0; it < 10; ++it) {
        iot_matvec_recip_kernel<<<S, 256, 0, stream>>>(Km, pb, pt);  // t  = 1/(K b)
        iot_matvec_recip_kernel<<<S, 256, 0, stream>>>(Km, pt, pa);  // a' = 1/(K t)  (new a after swap)
        float* tmp = pb; pb = pt; pt = tmp;                          // b' = t
    }
    iot_loss_kernel<<<1, 1024, 0, stream>>>(Km, pa, pb, (float*)d_out);
}
